// MSGAGNNLayer_78374563217886
// MI455X (gfx1250) — matrix operand order repaired and checked by reading, not compiled, not run on hardware
//
#include <hip/hip_runtime.h>
#include <hip/hip_bf16.h>
#include <math.h>

#define IN_DIM   64
#define N_ENT    40000
#define N_TOPK   256
#define BATCH    32
#define N_NODES  400000
#define N_PREV   100000
#define N_EDGES  500000

typedef __attribute__((ext_vector_type(16))) __bf16 v16bf;
typedef __attribute__((ext_vector_type(8)))  float  v8f;

__device__ __forceinline__ v8f wmma_bf16(v16bf a, v16bf b, v8f c) {
  // D = A(16x32 bf16) x B(32x16 bf16) + C(16x16 f32)
  return __builtin_amdgcn_wmma_f32_16x16x32_bf16(false, a, false, b, (short)0, c, false, false);
}

// monotone float -> uint key (order preserving, -inf smallest)
__device__ __forceinline__ unsigned fkey(float x) {
  unsigned u = __float_as_uint(x);
  return (u & 0x80000000u) ? ~u : (u | 0x80000000u);
}

// Store 4 consecutive-K bf16 values of row m into the swizzled A-fragment layout.
// Layout: chunk c (32 K), element (lane = hf*16 + m, j) at [(c*32+lane)*16 + j].
// K->(hf, j) per 16-bit A-matrix ISA layout: hf = (kl>>3)&1, j = (kl&7) + ((kl&16)>>1).
__device__ __forceinline__ void stash_bf4(__bf16* base, int m, int kabs, float4 v) {
  int c  = kabs >> 5, kl = kabs & 31;
  int hf = (kl >> 3) & 1;
  int j  = (kl & 7) + ((kl & 16) >> 1);
  __bf16* d = base + ((c * 32 + hf * 16 + m) * 16 + j);
  d[0] = (__bf16)v.x; d[1] = (__bf16)v.y; d[2] = (__bf16)v.z; d[3] = (__bf16)v.w;
}

// ---------------- prep kernels ----------------
__global__ void init_scores_kernel(float* __restrict__ ns) {
  int i = blockIdx.x * blockDim.x + threadIdx.x;
  if (i < BATCH * N_ENT) ns[i] = -INFINITY;
}

__global__ void mark_old_kernel(const int* __restrict__ old_idx, int* __restrict__ is_old) {
  int i = blockIdx.x * blockDim.x + threadIdx.x;
  if (i < N_PREV) is_old[old_idx[i]] = 1;
}

// Convert weights to bf16, pre-swizzled into the B-fragment layout:
// out[((c*4 + nt)*32 + lane)*16 + j] = W[k*64 + n],
//   k = c*32 + (lane>>4)*8 + (j&7) + (j>>3)*16, n = nt*16 + (lane&15).
__global__ void swizzle_weights_kernel(const float* __restrict__ Wa1, const float* __restrict__ Wlog,
                                       const float* __restrict__ Wmsg,
                                       __bf16* __restrict__ Wa1s, __bf16* __restrict__ Wlogs,
                                       __bf16* __restrict__ Wmsgs) {
  int i = blockIdx.x * blockDim.x + threadIdx.x;
  const float* src; __bf16* dst; int idx;
  if (i < 12288)            { src = Wa1;  dst = Wa1s;  idx = i; }
  else if (i < 12288+8192)  { src = Wlog; dst = Wlogs; idx = i - 12288; }
  else if (i < 12288+8192+4096) { src = Wmsg; dst = Wmsgs; idx = i - 12288 - 8192; }
  else return;
  int j = idx & 15, lane = (idx >> 4) & 31, f = idx >> 9;
  int nt = f & 3, c = f >> 2;
  int k = c * 32 + (lane >> 4) * 8 + (j & 7) + ((j >> 3) << 4);
  int n = nt * 16 + (lane & 15);
  dst[idx] = (__bf16)src[k * 64 + n];
}

// ---------------- K2: edge attention + message scatter ----------------
// 2 waves/block, one 16-edge tile per wave. A = [hs|hr|hqr] (16x192), B = Wa1 (192x64).
__global__ __launch_bounds__(64) void edge_kernel(
    const float* __restrict__ q_rel, const float* __restrict__ h_prev,
    const int* __restrict__ edges, const float* __restrict__ rela,
    const __bf16* __restrict__ Wa1s, const float* __restrict__ ba1,
    const float* __restrict__ Wa2, const float* __restrict__ ba2,
    float* __restrict__ h_agg) {
  __shared__ __align__(32) __bf16 sW[192 * 64];          // 24 KB swizzled B frags
  __shared__ __align__(32) __bf16 sAf[2][6 * 32 * 16];   // 12 KB swizzled A frags
  __shared__ float  sHs[2][16 * 64];                     //  8 KB
  __shared__ float  sHr[2][16 * 64];                     //  8 KB
  __shared__ float  sH1[2][16 * 64];                     //  8 KB
  __shared__ float  sAlpha[2][16];
  __shared__ int    sSub[2][16], sRel[2][16], sObj[2][16], sB[2][16];

  const int tid = threadIdx.x;
  const int w = tid >> 5, lane = tid & 31;

  __builtin_prefetch(Wa1s, 0, 3);
  for (int i = tid; i < 192 * 64; i += 64) sW[i] = Wa1s[i];

  const int base = (blockIdx.x * 2 + w) * 16;
  if (lane < 16) {
    int e = base + lane;
    sSub[w][lane] = edges[e * 6 + 4];
    sRel[w][lane] = edges[e * 6 + 2];
    sObj[w][lane] = edges[e * 6 + 5];
    sB[w][lane]   = edges[e * 6 + 0];
  }
  __syncthreads();

  // gather + convert + swizzle-stage: hs (K 0..63), hr (K 64..127), hqr (K 128..191)
  for (int t = lane; t < 256; t += 32) {
    int row = t >> 4, q = t & 15;
    float4 v = ((const float4*)h_prev)[(size_t)sSub[w][row] * 16 + q];
    ((float4*)&sHs[w][row * 64])[q] = v;
    stash_bf4(sAf[w], row, q * 4, v);
  }
  for (int t = lane; t < 256; t += 32) {
    int row = t >> 4, q = t & 15;
    float4 v = ((const float4*)rela)[(size_t)sRel[w][row] * 16 + q];
    ((float4*)&sHr[w][row * 64])[q] = v;
    stash_bf4(sAf[w], row, 64 + q * 4, v);
  }
  for (int t = lane; t < 256; t += 32) {
    int row = t >> 4, q = t & 15;
    float4 v = ((const float4*)q_rel)[(size_t)sB[w][row] * 16 + q];
    stash_bf4(sAf[w], row, 128 + q * 4, v);
  }
  __syncthreads();

  const int m = lane & 15, hf = lane >> 4;
  for (int nt = 0; nt < 4; ++nt) {
    const int n = nt * 16 + m;
    float bias = ba1[n];
    v8f acc = {bias, bias, bias, bias, bias, bias, bias, bias};
#pragma unroll
    for (int c = 0; c < 6; ++c) {
      v16bf a = *(const v16bf*)&sAf[w][(c * 32 + lane) * 16];
      v16bf b = *(const v16bf*)&sW[((c * 4 + nt) * 32 + lane) * 16];
      acc = wmma_bf16(a, b, acc);
    }
#pragma unroll
    for (int r = 0; r < 8; ++r)
      sH1[w][(r + hf * 8) * 64 + n] = fmaxf(acc[r], 0.0f);
  }
  __syncthreads();

  if (lane < 16) {
    float dot = ba2[0];
    for (int nn = 0; nn < 64; ++nn) dot += sH1[w][lane * 64 + nn] * Wa2[nn];
    sAlpha[w][lane] = 1.0f / (1.0f + __expf(-dot));
  }
  __syncthreads();

  for (int t = lane; t < 16 * 64; t += 32) {
    int e = t >> 6;
    float msg = sAlpha[w][e] * (sHs[w][t] + sHr[w][t]);
    atomicAdd(&h_agg[(size_t)sObj[w][e] * 64 + (t & 63)], msg);
  }
}

// ---------------- K3: h_old = relu([h_prev | h_agg[old]] @ Wlog + blog), in-place ----------------
__global__ __launch_bounds__(64) void hold_kernel(
    const float* __restrict__ h_prev, const int* __restrict__ old_idx,
    const __bf16* __restrict__ Wlogs, const float* __restrict__ blog,
    float* __restrict__ h_agg) {
  __shared__ __align__(32) __bf16 sW[128 * 64];          // 16 KB swizzled
  __shared__ __align__(32) __bf16 sAf[2][4 * 32 * 16];   //  8 KB swizzled
  __shared__ int sIdx[2][16];

  const int tid = threadIdx.x, w = tid >> 5, lane = tid & 31;
  __builtin_prefetch(Wlogs, 0, 3);
  for (int i = tid; i < 128 * 64; i += 64) sW[i] = Wlogs[i];
  const int base = (blockIdx.x * 2 + w) * 16;
  if (lane < 16) sIdx[w][lane] = old_idx[base + lane];
  __syncthreads();

  for (int t = lane; t < 256; t += 32) {      // h_prev rows (K 0..63), contiguous
    int row = t >> 4, q = t & 15;
    float4 v = ((const float4*)h_prev)[(size_t)(base + row) * 16 + q];
    stash_bf4(sAf[w], row, q * 4, v);
  }
  for (int t = lane; t < 256; t += 32) {      // gathered h_agg rows (K 64..127)
    int row = t >> 4, q = t & 15;
    float4 v = ((const float4*)h_agg)[(size_t)sIdx[w][row] * 16 + q];
    stash_bf4(sAf[w], row, 64 + q * 4, v);
  }
  __syncthreads();

  const int m = lane & 15, hf = lane >> 4;
  for (int nt = 0; nt < 4; ++nt) {
    const int n = nt * 16 + m;
    float bias = blog[n];
    v8f acc = {bias, bias, bias, bias, bias, bias, bias, bias};
#pragma unroll
    for (int c = 0; c < 4; ++c) {
      v16bf a = *(const v16bf*)&sAf[w][(c * 32 + lane) * 16];
      v16bf b = *(const v16bf*)&sW[((c * 4 + nt) * 32 + lane) * 16];
      acc = wmma_bf16(a, b, acc);
    }
#pragma unroll
    for (int r = 0; r < 8; ++r)
      h_agg[(size_t)sIdx[w][r + hf * 8] * 64 + n] = fmaxf(acc[r], 0.0f);
  }
}

// ---------------- K4: hidden_new = relu(h_logic @ Wmsg); scores; node_scores scatter ----------------
__global__ __launch_bounds__(128) void hidden_kernel(
    const float* __restrict__ h_agg, const __bf16* __restrict__ Wmsgs,
    const float* __restrict__ Wsamp, const float* __restrict__ bsamp,
    const int* __restrict__ nodes, const int* __restrict__ is_old,
    float* __restrict__ out_hidden, float* __restrict__ node_scores) {
  __shared__ __align__(32) __bf16 sW[64 * 64];           //  8 KB swizzled
  __shared__ __align__(32) __bf16 sAf[4][2 * 32 * 16];   //  8 KB swizzled
  __shared__ float sHN[4][16 * 64];                      // 16 KB

  const int tid = threadIdx.x, w = tid >> 5, lane = tid & 31;
  __builtin_prefetch(Wmsgs, 0, 3);
  for (int i = tid; i < 64 * 64; i += 128) sW[i] = Wmsgs[i];
  const int base = (blockIdx.x * 4 + w) * 16;

  for (int t = lane; t < 256; t += 32) {
    int row = t >> 4, q = t & 15;
    float4 v = ((const float4*)h_agg)[(size_t)(base + row) * 16 + q];
    stash_bf4(sAf[w], row, q * 4, v);
  }
  __syncthreads();

  const int m = lane & 15, hf = lane >> 4;
  for (int nt = 0; nt < 4; ++nt) {
    const int n = nt * 16 + m;
    v8f acc = {0.f, 0.f, 0.f, 0.f, 0.f, 0.f, 0.f, 0.f};
#pragma unroll
    for (int c = 0; c < 2; ++c) {
      v16bf a = *(const v16bf*)&sAf[w][(c * 32 + lane) * 16];
      v16bf b = *(const v16bf*)&sW[((c * 4 + nt) * 32 + lane) * 16];
      acc = wmma_bf16(a, b, acc);
    }
#pragma unroll
    for (int r = 0; r < 8; ++r) {
      int mm = r + hf * 8;
      float v = fmaxf(acc[r], 0.0f);
      out_hidden[(size_t)(base + mm) * 64 + n] = v;
      sHN[w][mm * 64 + n] = v;
    }
  }
  __syncthreads();

  if (lane < 16) {
    int i = base + lane;
    float dot = bsamp[0];
    for (int nn = 0; nn < 64; ++nn) dot += sHN[w][lane * 64 + nn] * Wsamp[nn];
    if (!is_old[i]) {
      int b = nodes[i * 2 + 0], ent = nodes[i * 2 + 1];
      node_scores[(size_t)b * N_ENT + ent] = dot;
    }
  }
}

// ---------------- K5: per-row radix-select of the 256th-largest score key ----------------
__global__ __launch_bounds__(256) void topk_kernel(const float* __restrict__ ns,
                                                   unsigned* __restrict__ thresh) {
  __shared__ unsigned hist[256];
  __shared__ unsigned sPrefix;
  __shared__ int      sRem;
  const int b = blockIdx.x, tid = threadIdx.x;
  const float* row = ns + (size_t)b * N_ENT;

  unsigned prefix = 0;
  int rem = N_TOPK;
  for (int level = 3; level >= 0; --level) {
    hist[tid] = 0;
    __syncthreads();
    unsigned hmask = (level == 3) ? 0u : (0xFFFFFFFFu << ((level + 1) * 8));
    for (int i = tid; i < N_ENT; i += 256) {
      unsigned key = fkey(row[i]);
      if ((key & hmask) == (prefix & hmask))
        atomicAdd(&hist[(key >> (level * 8)) & 255u], 1u);
    }
    __syncthreads();
    if (tid == 0) {
      int acc = 0;
      unsigned bin = 255;
      for (;; --bin) {
        int c = (int)hist[bin];
        if (acc + c >= rem || bin == 0) {
          sPrefix = prefix | (bin << (level * 8));
          sRem = rem - acc;
          break;
        }
        acc += c;
      }
    }
    __syncthreads();
    prefix = sPrefix;
    rem = sRem;
    __syncthreads();
  }
  if (tid == 0) thresh[b] = prefix;
}

// ---------------- K6: final mask + zero non-selected rows ----------------
__global__ __launch_bounds__(256) void finalize_kernel(
    const int* __restrict__ nodes, const int* __restrict__ is_old,
    const float* __restrict__ node_scores, const unsigned* __restrict__ thresh,
    float* __restrict__ out_hidden, float* __restrict__ out_mask) {
  int i = blockIdx.x * blockDim.x + threadIdx.x;
  if (i >= N_NODES) return;
  bool keep;
  if (is_old[i]) {
    keep = true;
  } else {
    int b = nodes[i * 2 + 0], ent = nodes[i * 2 + 1];
    keep = fkey(node_scores[(size_t)b * N_ENT + ent]) >= thresh[b];
  }
  out_mask[i] = keep ? 1.0f : 0.0f;
  if (!keep) {
    float4 z = {0.f, 0.f, 0.f, 0.f};
    float4* p = (float4*)(out_hidden + (size_t)i * 64);
#pragma unroll
    for (int q = 0; q < 16; ++q) p[q] = z;
  }
}

extern "C" void kernel_launch(void* const* d_in, const int* in_sizes, int n_in,
                              void* d_out, int out_size, void* d_ws, size_t ws_size,
                              hipStream_t stream) {
  (void)in_sizes; (void)n_in; (void)out_size; (void)ws_size;
  const float* q_rel   = (const float*)d_in[0];
  const float* h_prev  = (const float*)d_in[1];
  const int*   edges   = (const int*)d_in[2];
  const int*   nodes   = (const int*)d_in[3];
  const int*   old_idx = (const int*)d_in[4];
  const float* rela    = (const float*)d_in[5];
  const float* Wa1     = (const float*)d_in[6];
  const float* ba1     = (const float*)d_in[7];
  const float* Wa2     = (const float*)d_in[8];
  const float* ba2     = (const float*)d_in[9];
  const float* Wlog    = (const float*)d_in[10];
  const float* blog    = (const float*)d_in[11];
  const float* Wmsg    = (const float*)d_in[12];
  const float* Wsamp   = (const float*)d_in[13];
  const float* bsamp   = (const float*)d_in[14];

  char* ws = (char*)d_ws;
  size_t off = 0;
  auto alloc = [&](size_t bytes) -> void* {
    void* p = ws + off;
    off = (off + bytes + 255) & ~(size_t)255;
    return p;
  };
  float*    h_agg       = (float*)alloc((size_t)N_NODES * 64 * sizeof(float));
  float*    node_scores = (float*)alloc((size_t)BATCH * N_ENT * sizeof(float));
  int*      is_old      = (int*)alloc((size_t)N_NODES * sizeof(int));
  unsigned* thresh      = (unsigned*)alloc(BATCH * sizeof(unsigned));
  __bf16*   Wa1s        = (__bf16*)alloc(192 * 64 * sizeof(__bf16));
  __bf16*   Wlogs       = (__bf16*)alloc(128 * 64 * sizeof(__bf16));
  __bf16*   Wmsgs       = (__bf16*)alloc(64 * 64 * sizeof(__bf16));

  hipMemsetAsync(h_agg, 0, (size_t)N_NODES * 64 * sizeof(float), stream);
  hipMemsetAsync(is_old, 0, (size_t)N_NODES * sizeof(int), stream);
  init_scores_kernel<<<(BATCH * N_ENT + 255) / 256, 256, 0, stream>>>(node_scores);
  mark_old_kernel<<<(N_PREV + 255) / 256, 256, 0, stream>>>(old_idx, is_old);
  swizzle_weights_kernel<<<(12288 + 8192 + 4096 + 255) / 256, 256, 0, stream>>>(
      Wa1, Wlog, Wmsg, Wa1s, Wlogs, Wmsgs);

  edge_kernel<<<N_EDGES / 32, 64, 0, stream>>>(q_rel, h_prev, edges, rela, Wa1s, ba1, Wa2, ba2, h_agg);
  hold_kernel<<<N_PREV / 32, 64, 0, stream>>>(h_prev, old_idx, Wlogs, blog, h_agg);

  float* out_hidden = (float*)d_out;
  float* out_mask   = out_hidden + (size_t)N_NODES * 64;
  hidden_kernel<<<N_NODES / 64, 128, 0, stream>>>(h_agg, Wmsgs, Wsamp, bsamp, nodes, is_old,
                                                  out_hidden, node_scores);
  topk_kernel<<<BATCH, 256, 0, stream>>>(node_scores, thresh);
  finalize_kernel<<<(N_NODES + 255) / 256, 256, 0, stream>>>(nodes, is_old, node_scores, thresh,
                                                             out_hidden, out_mask);
}
